// ResidualTransformerRegressionModule_30021821399849
// MI455X (gfx1250) — matrix operand order repaired and checked by reading, not compiled, not run on hardware
//
#include <hip/hip_runtime.h>

#define NPTS   8192
#define PB     1024
#define NB     8
#define KNBR   20
#define HDIM   256
#define HHID   512
#define NLAY   3
#define NCOMP  512
#define LN_EPS 1e-5f

typedef __attribute__((ext_vector_type(16))) _Float16 v16h;
typedef __attribute__((ext_vector_type(8)))  float    v8f;
typedef __attribute__((ext_vector_type(4)))  float    v4f;
typedef __attribute__((ext_vector_type(4)))  unsigned int u32x4;
typedef __attribute__((ext_vector_type(8)))  int      i32x8;
typedef __attribute__((ext_vector_type(4)))  int      i32x4;

union HV { v16h h16; v4f f4[2]; };

__device__ __forceinline__ float wave_sum(float v) {
#pragma unroll
  for (int o = 16; o > 0; o >>= 1) v += __shfl_xor(v, o, 32);
  return v;
}

__device__ __forceinline__ float elu1(float v) { return v > 0.f ? v : __expf(v) - 1.f; }

// --------------------------------------------------------- TDM row-tile load -
// Copy nrows contiguous rows of row_bytes each from global into LDS via the
// Tensor Data Mover (D# per CDNA5 ISA ch.8; data_size=1B, 2-D tile).
__device__ __forceinline__ void tdm_load_rows_to_lds(const void* gaddr, unsigned lds_off,
                                                     unsigned row_bytes, unsigned nrows,
                                                     unsigned tensor_rows) {
  unsigned long long ga = (unsigned long long)(size_t)gaddr;
  u32x4 g0;
  g0[0] = 1u;                                        // count=1 (valid user D#)
  g0[1] = lds_off;                                   // lds_addr (bytes)
  g0[2] = (unsigned)(ga & 0xffffffffu);              // global_addr[31:0]
  g0[3] = (unsigned)((ga >> 32) & 0x01ffffffu)       // global_addr[56:32]
          | (2u << 30);                              // type = 2 ("image")
  i32x8 g1;
  g1[0] = 0;                                         // mask=0, data_size=0(1B)
  g1[1] = (int)(row_bytes << 16);                    // tensor_dim0[15:0]
  g1[2] = (int)(tensor_rows << 16);                  // dim0 hi | tensor_dim1 lo
  g1[3] = (int)(row_bytes << 16);                    // dim1 hi | tile_dim0
  g1[4] = (int)nrows;                                // tile_dim1 | tile_dim2=0
  g1[5] = (int)row_bytes;                            // tensor_dim0_stride lo
  g1[6] = 0;
  g1[7] = 0;
  i32x4 gz = {0, 0, 0, 0};
#if __clang_major__ >= 23
  i32x8 gz8 = {0, 0, 0, 0, 0, 0, 0, 0};
  __builtin_amdgcn_tensor_load_to_lds(g0, g1, gz, gz, gz8, 0);
#else
  __builtin_amdgcn_tensor_load_to_lds(g0, g1, gz, gz, 0);
#endif
}

// ---------------------------------------------------------------- KNN --------
__global__ __launch_bounds__(256)
void knn_kernel(const float* __restrict__ pos, int* __restrict__ neigh) {
  int i = blockIdx.x * blockDim.x + threadIdx.x;
  if (i >= NPTS) return;
  int base = (i >> 10) << 10;  // cloud start (P=1024)
  float px = pos[i * 3], py = pos[i * 3 + 1], pz = pos[i * 3 + 2];
  float bd[KNBR]; int bj[KNBR];
#pragma unroll
  for (int t = 0; t < KNBR; ++t) { bd[t] = 3.0e38f; bj[t] = base; }
  for (int jj = 0; jj < PB; ++jj) {
    int j = base + jj;
    if (j == i) continue;
    float dx = px - pos[j * 3], dy = py - pos[j * 3 + 1], dz = pz - pos[j * 3 + 2];
    float d2 = dx * dx + dy * dy + dz * dz;
    if (d2 < bd[KNBR - 1]) {
      float cd = d2; int cj = j;
#pragma unroll
      for (int t = 0; t < KNBR; ++t) {
        if (cd < bd[t]) { float td = bd[t]; int tj = bj[t]; bd[t] = cd; bj[t] = cj; cd = td; cj = tj; }
      }
    }
  }
#pragma unroll
  for (int t = 0; t < KNBR; ++t) neigh[(size_t)i * KNBR + t] = bj[t];
}

// ---------------------------------------------------------------- stem -------
__global__ __launch_bounds__(256)
void stem_kernel(const float* __restrict__ pos, const float* __restrict__ W,
                 const float* __restrict__ b, float* __restrict__ x,
                 _Float16* __restrict__ xh) {
  int t = blockIdx.x * blockDim.x + threadIdx.x;
  if (t >= NPTS * HDIM) return;
  int i = t >> 8, c = t & 255;
  float v = b[c] + pos[i * 3] * W[c] + pos[i * 3 + 1] * W[HDIM + c] + pos[i * 3 + 2] * W[2 * HDIM + c];
  x[t] = v;
  xh[t] = (_Float16)v;
}

// --------------------------------------------- weight pack to B-fragments ----
// B tile (kt,nt) of a Kdim x Ndim row-major fp32 matrix -> 32 lanes x 16 halves.
// Lane l: column n = nt*16 + (l&15); k = kt*32 + (h&8)*2 + (l>>4)*8 + (h&7)  (h = 0..15).
__global__ __launch_bounds__(256)
void pack_w_kernel(const float* __restrict__ W, _Float16* __restrict__ out,
                   int Kdim, int Ndim) {
  int t = blockIdx.x * blockDim.x + threadIdx.x;
  int ktiles = Kdim >> 5, ntiles = Ndim >> 4;
  int total = ktiles * ntiles * 32;
  if (t >= total) return;
  int lane = t & 31, tile = t >> 5;
  int nt = tile % ntiles, kt = tile / ntiles;
  int n = nt * 16 + (lane & 15);
  int kbase = kt * 32 + (lane >> 4) * 8;
  _Float16* dst = out + (size_t)t * 16;
#pragma unroll
  for (int h = 0; h < 16; ++h) dst[h] = (_Float16)W[(size_t)(kbase + (h & 8) * 2 + (h & 7)) * Ndim + n];
}

// ------------------------------------------------- WMMA GEMM (LDS A, LDS out)
__device__ __forceinline__ void wmma_gemm_lds(
    const _Float16* __restrict__ A, int a_ld, int m_tiles, int m_valid,
    const _Float16* __restrict__ Bp, int ktiles, int ntiles,
    const float* __restrict__ bias,
    _Float16* __restrict__ Out, int o_ld, int wave, int nwaves, int lane) {
  const int hi = lane >> 4, hl = lane & 15;
  const int total = m_tiles * ntiles;
  for (int t = wave; t < total; t += nwaves) {
    const int mt = t / ntiles, nt = t - mt * ntiles;
    v8f acc = {};
    const _Float16* arow = A + (mt * 16 + hl) * a_ld + hi * 8;
    for (int kt = 0; kt < ktiles; ++kt) {
      HV a, b;
      const _Float16* ap = arow + kt * 32;       // A 16x32 f16 fragment: two 8-half chunks
      a.f4[0] = *(const v4f*)ap;
      a.f4[1] = *(const v4f*)(ap + 16);
      const _Float16* bp = Bp + ((size_t)(kt * ntiles + nt) * 32 + lane) * 16;
      b.f4[0] = *(const v4f*)bp;
      b.f4[1] = *(const v4f*)(bp + 8);
      if (kt + 1 < ktiles)
        __builtin_prefetch(Bp + ((size_t)((kt + 1) * ntiles + nt) * 32 + lane) * 16, 0, 0);
      acc = __builtin_amdgcn_wmma_f32_16x16x32_f16(false, a.h16, false, b.h16,
                                                   (short)0, acc, false, false);
    }
    const int n = nt * 16 + hl;
    const float bv = bias ? bias[n] : 0.f;
    const int rb = mt * 16 + hi * 8;             // D: lane holds col n, rows rb..rb+7
#pragma unroll
    for (int r = 0; r < 8; ++r) {
      if (rb + r < m_valid) Out[(rb + r) * o_ld + n] = (_Float16)(acc[r] + bv);
    }
  }
}

// ---------------------------------------------- row LayerNorm + ELU on LDS ---
__device__ __forceinline__ void ln_elu_h(_Float16* buf, int ld, int rows, int width,
                                         const float* __restrict__ g,
                                         const float* __restrict__ be,
                                         int wave, int nwaves, int lane) {
  for (int r = wave; r < rows; r += nwaves) {
    _Float16* row = buf + r * ld;
    float s = 0.f, s2 = 0.f;
    for (int c = lane; c < width; c += 32) { float v = (float)row[c]; s += v; s2 += v * v; }
    s = wave_sum(s); s2 = wave_sum(s2);
    float inv = 1.f / (float)width;
    float m = s * inv, var = s2 * inv - m * m;
    float rs = rsqrtf(var + LN_EPS);
    for (int c = lane; c < width; c += 32) {
      float v = ((float)row[c] - m) * rs * g[c] + be[c];
      row[c] = (_Float16)elu1(v);
    }
  }
}

// ------------------------------------------------- node GEMM (xh @ W, WMMA) --
// TDM copies the 64x256 fp16 node-feature tile into LDS (one DMA per block),
// then 4 waves run 16 WMMA output tiles each. Two instantiations avoid any
// per-store branching: F32OUT=false -> f16 out, no bias; true -> f32 out + bias.
template <bool F32OUT>
__global__ __launch_bounds__(128)
void node_gemm_kernel(const _Float16* __restrict__ Xh, const _Float16* __restrict__ Bp,
                      const float* __restrict__ bias,
                      _Float16* __restrict__ out_h, float* __restrict__ out_f) {
  __shared__ _Float16 xt[64 * HDIM];
  int tid = threadIdx.x, lane = tid & 31, wave = tid >> 5;
  int row0 = blockIdx.x * 64;
  if (tid == 0) {
    unsigned lds_off = (unsigned)(size_t)(&xt[0]);   // addr[31:0] == LDS offset
    tdm_load_rows_to_lds(Xh + (size_t)row0 * HDIM, lds_off,
                         HDIM * 2 /*bytes/row*/, 64 /*rows*/, NPTS);
    __builtin_amdgcn_s_wait_tensorcnt(0);
  }
  __syncthreads();
  const int hi = lane >> 4, hl = lane & 15;
  const int ktiles = HDIM / 32, ntiles = HDIM / 16;  // 8, 16
  for (int t = wave; t < 4 * ntiles; t += 4) {
    int mt = t / ntiles, nt = t - mt * ntiles;
    v8f acc = {};
    for (int kt = 0; kt < ktiles; ++kt) {
      HV a, b;
      const _Float16* ap = xt + (mt * 16 + hl) * HDIM + kt * 32 + hi * 8;
      a.f4[0] = *(const v4f*)ap;
      a.f4[1] = *(const v4f*)(ap + 16);
      const _Float16* bp = Bp + ((size_t)(kt * ntiles + nt) * 32 + lane) * 16;
      b.f4[0] = *(const v4f*)bp;
      b.f4[1] = *(const v4f*)(bp + 8);
      acc = __builtin_amdgcn_wmma_f32_16x16x32_f16(false, a.h16, false, b.h16,
                                                   (short)0, acc, false, false);
    }
    int n = nt * 16 + hl;
    int rb = row0 + mt * 16 + hi * 8;
    if constexpr (F32OUT) {
      float bv = bias[n];
#pragma unroll
      for (int r = 0; r < 8; ++r) out_f[(size_t)(rb + r) * HDIM + n] = acc[r] + bv;
    } else {
#pragma unroll
      for (int r = 0; r < 8; ++r) out_h[(size_t)(rb + r) * HDIM + n] = (_Float16)acc[r];
    }
  }
}

// ------------------------------------ fused per-node edge pipeline (1 layer) -
__global__ __launch_bounds__(128)
void edge_layer_kernel(
    const float* __restrict__ pos, const int* __restrict__ neigh,
    const float* __restrict__ x_in, float* __restrict__ x_out,
    _Float16* __restrict__ x_out_h,
    const _Float16* __restrict__ aSrc, const _Float16* __restrict__ aDst,
    const _Float16* __restrict__ vH,
    const float* __restrict__ pos1_W, const float* __restrict__ pos1_b,
    const float* __restrict__ pln1_g, const float* __restrict__ pln1_b,
    const _Float16* __restrict__ pwPos2, const float* __restrict__ pos2_b,
    const float* __restrict__ pln2_g, const float* __restrict__ pln2_b,
    const _Float16* __restrict__ pwAtt1, const float* __restrict__ att1_b,
    const float* __restrict__ aln1_g, const float* __restrict__ aln1_b,
    const _Float16* __restrict__ pwAtt2, const float* __restrict__ att2_b,
    const float* __restrict__ aln2_g, const float* __restrict__ aln2_b,
    const float* __restrict__ blk_g, const float* __restrict__ blk_bt,
    const float* __restrict__ blk_W, const float* __restrict__ blk_b,
    const float* __restrict__ res_alpha, int layer) {
  const int i = blockIdx.x;
  const int tid = threadIdx.x, lane = tid & 31, wave = tid >> 5;

  extern __shared__ char smem[];
  _Float16* bufA = (_Float16*)smem;            // 32x512 f16 (h1, later t)
  _Float16* bufB = bufA + 32 * HHID;           // 32x256 f16 (delta)
  _Float16* bufC = bufB + 32 * HDIM;           // 32x256 f16 (s, later alpha)
  // scratch hidden in bufC rows 20..31 (never touched by guarded GEMM stores)
  float* aggS = (float*)(bufC + KNBR * HDIM);  // 256 f
  float* yS   = aggS + HDIM;                   // 256 f
  float* relS = yS + HDIM;                     // 64 f (20x3)
  int*   nIdx = (int*)(relS + 64);             // 20 i

  // 1) neighbor ids + relative positions
  if (tid < KNBR) {
    int j = neigh[(size_t)i * KNBR + tid];
    nIdx[tid] = j;
    relS[tid * 3 + 0] = pos[i * 3 + 0] - pos[j * 3 + 0];
    relS[tid * 3 + 1] = pos[i * 3 + 1] - pos[j * 3 + 1];
    relS[tid * 3 + 2] = pos[i * 3 + 2] - pos[j * 3 + 2];
  }
  __syncthreads();

  // 2) pos_nn layer-1 (K=3 -> scalar): raw h1 into bufA
  for (int e = tid; e < KNBR * HHID; e += 128) {
    int r = e >> 9, c = e & 511;
    float val = pos1_b[c] + relS[r * 3] * pos1_W[c] +
                relS[r * 3 + 1] * pos1_W[HHID + c] +
                relS[r * 3 + 2] * pos1_W[2 * HHID + c];
    bufA[r * HHID + c] = (_Float16)val;
  }
  __syncthreads();
  ln_elu_h(bufA, HHID, KNBR, HHID, pln1_g, pln1_b, wave, 4, lane);
  __syncthreads();

  // 3) delta = ELU(LN(h1 @ pos2_W + b))  -- WMMA 32x512x256
  wmma_gemm_lds(bufA, HHID, 2, KNBR, pwPos2, HHID / 32, HDIM / 16, pos2_b, bufB, HDIM, wave, 4, lane);
  __syncthreads();
  ln_elu_h(bufB, HDIM, KNBR, HDIM, pln2_g, pln2_b, wave, 4, lane);
  __syncthreads();

  // 4) s = a_dst[i] - a_src[j] + delta -> bufC (f16)
  for (int e = tid; e < KNBR * HDIM; e += 128) {
    int r = e >> 8, c = e & 255;
    float sv = (float)aDst[(size_t)i * HDIM + c] -
               (float)aSrc[(size_t)nIdx[r] * HDIM + c] + (float)bufB[r * HDIM + c];
    bufC[r * HDIM + c] = (_Float16)sv;
  }
  __syncthreads();

  // 5) t = ELU(LN(s @ att1_W + b))  -- WMMA 32x256x512 -> bufA
  wmma_gemm_lds(bufC, HDIM, 2, KNBR, pwAtt1, HDIM / 32, HHID / 16, att1_b, bufA, HHID, wave, 4, lane);
  __syncthreads();
  ln_elu_h(bufA, HHID, KNBR, HHID, aln1_g, aln1_b, wave, 4, lane);
  __syncthreads();

  // 6) alpha = ELU(LN(t @ att2_W + b)) -- WMMA 32x512x256 -> bufC
  wmma_gemm_lds(bufA, HHID, 2, KNBR, pwAtt2, HHID / 32, HDIM / 16, att2_b, bufC, HDIM, wave, 4, lane);
  __syncthreads();
  ln_elu_h(bufC, HDIM, KNBR, HDIM, aln2_g, aln2_b, wave, 4, lane);
  __syncthreads();

  // 7) per-channel softmax over K, aggregate attn * (v[j] + delta)
  for (int c = tid; c < HDIM; c += 128) {
    float mx = -3.0e38f;
    for (int r = 0; r < KNBR; ++r) mx = fmaxf(mx, (float)bufC[r * HDIM + c]);
    float ex[KNBR]; float ssum = 0.f;
#pragma unroll
    for (int r = 0; r < KNBR; ++r) { float e = __expf((float)bufC[r * HDIM + c] - mx); ex[r] = e; ssum += e; }
    float rinv = 1.f / ssum, acc = 0.f;
#pragma unroll
    for (int r = 0; r < KNBR; ++r) {
      float vv = (float)vH[(size_t)nIdx[r] * HDIM + c] + (float)bufB[r * HDIM + c];
      acc += ex[r] * rinv * vv;
    }
    aggS[c] = acc;
  }
  __syncthreads();

  // 8) y = ELU(LN(agg))  (wave 0)
  if (wave == 0) {
    float s = 0.f, s2 = 0.f;
    for (int c = lane; c < HDIM; c += 32) { float v = aggS[c]; s += v; s2 += v * v; }
    s = wave_sum(s); s2 = wave_sum(s2);
    float m = s / HDIM, var = s2 / HDIM - m * m;
    float rs = rsqrtf(var + LN_EPS);
    for (int c = lane; c < HDIM; c += 32) {
      float v = (aggS[c] - m) * rs * blk_g[c] + blk_bt[c];
      yS[c] = elu1(v);
    }
  }
  __syncthreads();

  // 9) h = y @ blk_W + b ; x_out = x_in + res_alpha * h  (f32 + f16 mirror)
  float ra = res_alpha[layer];
  for (int c = tid; c < HDIM; c += 128) {
    float acc = blk_b[c];
    for (int k = 0; k < HDIM; ++k) acc += yS[k] * blk_W[(size_t)k * HDIM + c];
    float vfin = x_in[(size_t)i * HDIM + c] + ra * acc;
    x_out[(size_t)i * HDIM + c] = vfin;
    x_out_h[(size_t)i * HDIM + c] = (_Float16)vfin;
  }
}

// ---------------------------------------------------------------- pooling ----
__global__ __launch_bounds__(256)
void segmax_kernel(const float* __restrict__ h, float* __restrict__ pooled) {
  int b = blockIdx.x, c = threadIdx.x;
  float m = -3.0e38f;
  for (int p = 0; p < PB; ++p) m = fmaxf(m, h[((size_t)b * PB + p) * HDIM + c]);
  pooled[b * HDIM + c] = m;
}

// ---------------------------------------------------------------- head tail --
__global__ __launch_bounds__(256)
void head_tail_kernel(const float* __restrict__ pooled,
                      const float* __restrict__ r2W, const float* __restrict__ r2b,
                      const float* __restrict__ g2, const float* __restrict__ be2,
                      const float* __restrict__ r3W, const float* __restrict__ r3b,
                      const float* __restrict__ g3, const float* __restrict__ be3,
                      const float* __restrict__ r4W, const float* __restrict__ r4b,
                      float* __restrict__ out) {
  __shared__ float p0[NB * HDIM], t1[NB * HDIM], t2[NB * HDIM];
  int tid = threadIdx.x, lane = tid & 31, wave = tid >> 5;  // 8 waves == 8 rows
  for (int e = tid; e < NB * HDIM; e += 256) p0[e] = pooled[e];
  __syncthreads();
  for (int e = tid; e < NB * HDIM; e += 256) {
    int r = e >> 8, c = e & 255;
    float a = r2b[c];
    for (int k = 0; k < HDIM; ++k) a += p0[r * HDIM + k] * r2W[(size_t)k * HDIM + c];
    t1[e] = a;
  }
  __syncthreads();
  {
    int r = wave;
    float s = 0.f, s2 = 0.f;
    for (int c = lane; c < HDIM; c += 32) { float v = t1[r * HDIM + c]; s += v; s2 += v * v; }
    s = wave_sum(s); s2 = wave_sum(s2);
    float m = s / HDIM, var = s2 / HDIM - m * m, rs = rsqrtf(var + LN_EPS);
    for (int c = lane; c < HDIM; c += 32)
      t1[r * HDIM + c] = elu1((t1[r * HDIM + c] - m) * rs * g2[c] + be2[c]);
  }
  __syncthreads();
  for (int e = tid; e < NB * HDIM; e += 256) {
    int r = e >> 8, c = e & 255;
    float a = r3b[c];
    for (int k = 0; k < HDIM; ++k) a += t1[r * HDIM + k] * r3W[(size_t)k * HDIM + c];
    t2[e] = a;
  }
  __syncthreads();
  {
    int r = wave;
    float s = 0.f, s2 = 0.f;
    for (int c = lane; c < HDIM; c += 32) { float v = t2[r * HDIM + c]; s += v; s2 += v * v; }
    s = wave_sum(s); s2 = wave_sum(s2);
    float m = s / HDIM, var = s2 / HDIM - m * m, rs = rsqrtf(var + LN_EPS);
    for (int c = lane; c < HDIM; c += 32)
      t2[r * HDIM + c] = elu1((t2[r * HDIM + c] - m) * rs * g3[c] + be3[c]);
  }
  __syncthreads();
  for (int e = tid; e < NB * NCOMP; e += 256) {
    int r = e >> 9, c = e & 511;
    float a = r4b[c];
    for (int k = 0; k < HDIM; ++k) a += t2[r * HDIM + k] * r4W[(size_t)k * NCOMP + c];
    out[e] = a;
  }
}

// ================================================================ launcher ===
extern "C" void kernel_launch(void* const* d_in, const int* in_sizes, int n_in,
                              void* d_out, int out_size, void* d_ws, size_t ws_size,
                              hipStream_t stream) {
  (void)in_sizes; (void)n_in; (void)out_size; (void)ws_size;
  const float* pos   = (const float*)d_in[0];
  const float* ffm_W = (const float*)d_in[2];  const float* ffm_b = (const float*)d_in[3];
  const float* lin_W = (const float*)d_in[4];  const float* src_W = (const float*)d_in[5];
  const float* dst_W = (const float*)d_in[6];
  const float* pos1_W = (const float*)d_in[7];  const float* pos1_b = (const float*)d_in[8];
  const float* pln1_g = (const float*)d_in[9];  const float* pln1_b = (const float*)d_in[10];
  const float* pos2_W = (const float*)d_in[11]; const float* pos2_b = (const float*)d_in[12];
  const float* pln2_g = (const float*)d_in[13]; const float* pln2_b = (const float*)d_in[14];
  const float* att1_W = (const float*)d_in[15]; const float* att1_b = (const float*)d_in[16];
  const float* aln1_g = (const float*)d_in[17]; const float* aln1_b = (const float*)d_in[18];
  const float* att2_W = (const float*)d_in[19]; const float* att2_b = (const float*)d_in[20];
  const float* aln2_g = (const float*)d_in[21]; const float* aln2_b = (const float*)d_in[22];
  const float* blk_g  = (const float*)d_in[23]; const float* blk_bt = (const float*)d_in[24];
  const float* blk_W  = (const float*)d_in[25]; const float* blk_b  = (const float*)d_in[26];
  const float* res_alpha = (const float*)d_in[27];
  const float* r1W = (const float*)d_in[28]; const float* r1b = (const float*)d_in[29];
  const float* r2W = (const float*)d_in[30]; const float* r2b = (const float*)d_in[31];
  const float* g2  = (const float*)d_in[32]; const float* be2 = (const float*)d_in[33];
  const float* r3W = (const float*)d_in[34]; const float* r3b = (const float*)d_in[35];
  const float* g3  = (const float*)d_in[36]; const float* be3 = (const float*)d_in[37];
  const float* r4W = (const float*)d_in[38]; const float* r4b = (const float*)d_in[39];

  char* ws = (char*)d_ws;
  size_t off = 0;
  auto take = [&](size_t bytes) -> char* {
    char* p = ws + off; off = (off + bytes + 255) & ~(size_t)255; return p;
  };
  int*      neigh  = (int*)take((size_t)NPTS * KNBR * 4);
  float*    x0     = (float*)take((size_t)NPTS * HDIM * 4);
  float*    x1     = (float*)take((size_t)NPTS * HDIM * 4);
  _Float16* x0h    = (_Float16*)take((size_t)NPTS * HDIM * 2);
  _Float16* x1h    = (_Float16*)take((size_t)NPTS * HDIM * 2);
  _Float16* aSrc   = (_Float16*)take((size_t)NPTS * HDIM * 2);
  _Float16* aDst   = (_Float16*)take((size_t)NPTS * HDIM * 2);
  _Float16* vHp    = (_Float16*)take((size_t)NPTS * HDIM * 2);
  float*    h1f    = (float*)take((size_t)NPTS * HDIM * 4);
  float*    pooled = (float*)take((size_t)NB * HDIM * 4);
  _Float16 *pwPos2[NLAY], *pwAtt1[NLAY], *pwAtt2[NLAY];
  _Float16 *pwSrc[NLAY], *pwDst[NLAY], *pwLin[NLAY];
  for (int l = 0; l < NLAY; ++l) {
    pwPos2[l] = (_Float16*)take((size_t)HHID * HDIM * 2);
    pwAtt1[l] = (_Float16*)take((size_t)HDIM * HHID * 2);
    pwAtt2[l] = (_Float16*)take((size_t)HHID * HDIM * 2);
    pwSrc[l]  = (_Float16*)take((size_t)HDIM * HDIM * 2);
    pwDst[l]  = (_Float16*)take((size_t)HDIM * HDIM * 2);
    pwLin[l]  = (_Float16*)take((size_t)HDIM * HDIM * 2);
  }
  _Float16* pwR1 = (_Float16*)take((size_t)HDIM * HDIM * 2);

  knn_kernel<<<NPTS / 256, 256, 0, stream>>>(pos, neigh);
  stem_kernel<<<(NPTS * HDIM) / 256, 256, 0, stream>>>(pos, ffm_W, ffm_b, x0, x0h);

  for (int l = 0; l < NLAY; ++l) {
    pack_w_kernel<<<32, 256, 0, stream>>>(pos2_W + (size_t)l * HHID * HDIM, pwPos2[l], HHID, HDIM);
    pack_w_kernel<<<32, 256, 0, stream>>>(att1_W + (size_t)l * HDIM * HHID, pwAtt1[l], HDIM, HHID);
    pack_w_kernel<<<32, 256, 0, stream>>>(att2_W + (size_t)l * HHID * HDIM, pwAtt2[l], HHID, HDIM);
    pack_w_kernel<<<16, 256, 0, stream>>>(src_W + (size_t)l * HDIM * HDIM, pwSrc[l], HDIM, HDIM);
    pack_w_kernel<<<16, 256, 0, stream>>>(dst_W + (size_t)l * HDIM * HDIM, pwDst[l], HDIM, HDIM);
    pack_w_kernel<<<16, 256, 0, stream>>>(lin_W + (size_t)l * HDIM * HDIM, pwLin[l], HDIM, HDIM);
  }
  pack_w_kernel<<<16, 256, 0, stream>>>(r1W, pwR1, HDIM, HDIM);

  float* xc = x0; float* xn = x1;
  _Float16* xch = x0h; _Float16* xnh = x1h;
  for (int l = 0; l < NLAY; ++l) {
    node_gemm_kernel<false><<<NPTS / 64, 128, 0, stream>>>(xch, pwSrc[l], nullptr, aSrc, nullptr);
    node_gemm_kernel<false><<<NPTS / 64, 128, 0, stream>>>(xch, pwDst[l], nullptr, aDst, nullptr);
    node_gemm_kernel<false><<<NPTS / 64, 128, 0, stream>>>(xch, pwLin[l], nullptr, vHp, nullptr);
    edge_layer_kernel<<<NPTS, 128, 65536, stream>>>(
        pos, neigh, xc, xn, xnh, aSrc, aDst, vHp,
        pos1_W + (size_t)l * 3 * HHID, pos1_b + (size_t)l * HHID,
        pln1_g + (size_t)l * HHID, pln1_b + (size_t)l * HHID,
        pwPos2[l], pos2_b + (size_t)l * HDIM,
        pln2_g + (size_t)l * HDIM, pln2_b + (size_t)l * HDIM,
        pwAtt1[l], att1_b + (size_t)l * HHID,
        aln1_g + (size_t)l * HHID, aln1_b + (size_t)l * HHID,
        pwAtt2[l], att2_b + (size_t)l * HDIM,
        aln2_g + (size_t)l * HDIM, aln2_b + (size_t)l * HDIM,
        blk_g + (size_t)l * HDIM, blk_bt + (size_t)l * HDIM,
        blk_W + (size_t)l * HDIM * HDIM, blk_b + (size_t)l * HDIM,
        res_alpha, l);
    float* tf = xc; xc = xn; xn = tf;
    _Float16* th = xch; xch = xnh; xnh = th;
  }

  node_gemm_kernel<true><<<NPTS / 64, 128, 0, stream>>>(xch, pwR1, r1b, nullptr, h1f);
  segmax_kernel<<<NB, 256, 0, stream>>>(h1f, pooled);
  head_tail_kernel<<<1, 256, 0, stream>>>(pooled, r2W, r2b, g2, be2,
                                          r3W, r3b, g3, be3, r4W, r4b, (float*)d_out);
}
